// SimCSE_16965120819485
// MI455X (gfx1250) — matrix operand order repaired and checked by reading, not compiled, not run on hardware
//
#include <hip/hip_runtime.h>
#include <hip/hip_bf16.h>

typedef __attribute__((ext_vector_type(16))) __bf16 v16bf;
typedef __attribute__((ext_vector_type(8)))  __bf16 v8bf;
typedef __attribute__((ext_vector_type(8)))  float  v8f;

#define SIMN 8192
#define SIMD 256
static constexpr float TEMP  = 0.05f;
static constexpr float EPSN  = 1e-8f;

// ---------------------------------------------------------------------------
// Kernel 1: per-row normalize, split fp32 -> bf16 hi/lo planes, emit diag.
// One block (256 threads) per row; each thread owns one of the D=256 elements.
// ---------------------------------------------------------------------------
__global__ void simcse_norm_split(const float* __restrict__ q,
                                  const float* __restrict__ p,
                                  __bf16* __restrict__ qhi, __bf16* __restrict__ qlo,
                                  __bf16* __restrict__ phi, __bf16* __restrict__ plo,
                                  float* __restrict__ diag) {
  const int row = blockIdx.x;
  const int t   = threadIdx.x;                 // 0..255 == K index
  const float qv = q[row * SIMD + t];
  const float pv = p[row * SIMD + t];
  float qq = qv * qv, pp = pv * pv, qp = qv * pv;
  #pragma unroll
  for (int off = 16; off > 0; off >>= 1) {     // wave32 reduce
    qq += __shfl_down(qq, off);
    pp += __shfl_down(pp, off);
    qp += __shfl_down(qp, off);
  }
  __shared__ float sqq[8], spp[8], sqp[8];
  const int wave = t >> 5, lane = t & 31;
  if (lane == 0) { sqq[wave] = qq; spp[wave] = pp; sqp[wave] = qp; }
  __syncthreads();
  float tqq = 0.f, tpp = 0.f, tqp = 0.f;
  #pragma unroll
  for (int w = 0; w < 8; ++w) { tqq += sqq[w]; tpp += spp[w]; tqp += sqp[w]; }

  const float rq = 1.0f / fmaxf(__builtin_sqrtf(tqq), EPSN);
  const float rp = 1.0f / fmaxf(__builtin_sqrtf(tpp), EPSN);
  const float qn = qv * rq, pn = pv * rp;
  const __bf16 qh = (__bf16)qn, ph = (__bf16)pn;
  qhi[row * SIMD + t] = qh;
  qlo[row * SIMD + t] = (__bf16)(qn - (float)qh);
  phi[row * SIMD + t] = ph;
  plo[row * SIMD + t] = (__bf16)(pn - (float)ph);
  if (t == 0) diag[row] = (tqp * rq * rp) / TEMP;   // sim_ii / T
}

// ---------------------------------------------------------------------------
// Kernel 2: fused GEMM (bf16x3 split-precision WMMA) + rowwise sum(exp(sim/T)).
// Block = 8 waves = 2 row-tiles x 4 column-phases over a 32-row stripe.
//   wave w: row-tile rt = w & 1  (rows r0 + rt*16 .. +15)
//           phase    ph = w >> 1 (column tiles j = ph, ph+4, ph+8, ...)
// The two row-tile waves of a phase stream identical B tiles -> WGP$ reuse,
// halving L2 traffic vs independent column sweeps. A fragments (hi+lo,
// full K=256) stay resident in VGPRs. Three independent WMMA accumulator
// chains (hh / hl / lh) keep the XDL pipe fed.
// ---------------------------------------------------------------------------
__global__ void __launch_bounds__(256, 1)
simcse_gemm_logz(const __bf16* __restrict__ qhi, const __bf16* __restrict__ qlo,
                 const __bf16* __restrict__ phi, const __bf16* __restrict__ plo,
                 float* __restrict__ logz) {
  const int wave  = threadIdx.x >> 5;
  const int lane  = threadIdx.x & 31;
  const int lhalf = lane >> 4;      // 0: lanes 0-15, 1: lanes 16-31
  const int lmod  = lane & 15;
  const int rt    = wave & 1;       // row tile within block
  const int ph    = wave >> 1;      // column phase (0..3)
  const int r0    = blockIdx.x * 32 + rt * 16;

  // ---- A fragments: ISA 16-bit A 16x32 layout.
  // lanes 0-15  hold row=lmod, K = k0 + {0..7, 16..23}
  // lanes 16-31 hold row=lmod, K = k0 + {8..15, 24..31}
  v16bf Ahi[8], Alo[8];
  {
    const int arow = r0 + lmod;
    const int koff = lhalf * 8;
    #pragma unroll
    for (int k = 0; k < 8; ++k) {
      const __bf16* bh = qhi + (size_t)arow * SIMD + k * 32 + koff;
      const __bf16* bl = qlo + (size_t)arow * SIMD + k * 32 + koff;
      v8bf h0 = *(const v8bf*)(bh);
      v8bf h1 = *(const v8bf*)(bh + 16);
      v8bf l0 = *(const v8bf*)(bl);
      v8bf l1 = *(const v8bf*)(bl + 16);
      Ahi[k] = __builtin_shufflevector(h0, h1, 0,1,2,3,4,5,6,7,8,9,10,11,12,13,14,15);
      Alo[k] = __builtin_shufflevector(l0, l1, 0,1,2,3,4,5,6,7,8,9,10,11,12,13,14,15);
    }
  }

  float rs[8] = {0.f,0.f,0.f,0.f,0.f,0.f,0.f,0.f};   // sum(exp) per C-row slot
  const float SCALE = 1.0f / TEMP;

  for (int j = ph; j < SIMN / 16; j += 4) {
    const int brow = j * 16 + lmod;
    const int bko  = lhalf * 8;    // B 32x16 layout: lanes0-15 K={0..7,16..23}, lanes16-31 K={8..15,24..31}
    // warm this wave's next column tile (global_prefetch_b8); over-run stays in ws
    __builtin_prefetch((const void*)(phi + (size_t)(brow + 64) * SIMD), 0, 1);

    v8f ahh = {}, ahl = {}, alh = {};   // 3 independent XDL chains
    #pragma unroll
    for (int k = 0; k < 8; ++k) {
      const __bf16* bh = phi + (size_t)brow * SIMD + k * 32 + bko;
      const __bf16* bl = plo + (size_t)brow * SIMD + k * 32 + bko;
      v8bf bh0 = *(const v8bf*)(bh);
      v8bf bh1 = *(const v8bf*)(bh + 16);
      v8bf bl0 = *(const v8bf*)(bl);
      v8bf bl1 = *(const v8bf*)(bl + 16);
      v16bf Bhi = __builtin_shufflevector(bh0, bh1, 0,1,2,3,4,5,6,7,8,9,10,11,12,13,14,15);
      v16bf Blo = __builtin_shufflevector(bl0, bl1, 0,1,2,3,4,5,6,7,8,9,10,11,12,13,14,15);
      // sim ~= hi*hi + hi*lo + lo*hi  (bf16x3 ~ fp32 accuracy)
      ahh = __builtin_amdgcn_wmma_f32_16x16x32_bf16(false, Ahi[k], false, Bhi,
                                                    (short)0, ahh, false, false);
      ahl = __builtin_amdgcn_wmma_f32_16x16x32_bf16(false, Ahi[k], false, Blo,
                                                    (short)0, ahl, false, false);
      alh = __builtin_amdgcn_wmma_f32_16x16x32_bf16(false, Alo[k], false, Bhi,
                                                    (short)0, alh, false, false);
    }
    // exp-sum epilogue; |sim/T| <= 20 so no max-shift needed in fp32.
    #pragma unroll
    for (int r = 0; r < 8; ++r)
      rs[r] += __expf(((ahh[r] + ahl[r]) + alh[r]) * SCALE);
  }

  // reduce the 16 column-lanes within each half (C layout: N = lane%16)
  #pragma unroll
  for (int r = 0; r < 8; ++r) {
    float v = rs[r];
    v += __shfl_xor(v, 1);
    v += __shfl_xor(v, 2);
    v += __shfl_xor(v, 4);
    v += __shfl_xor(v, 8);
    rs[r] = v;
  }
  // C layout: vgpr r, lanes0-15 -> M=r ; lanes16-31 -> M=8+r
  __shared__ float lds[8 * 16];
  if (lmod == 0) {
    #pragma unroll
    for (int r = 0; r < 8; ++r) lds[wave * 16 + lhalf * 8 + r] = rs[r];
  }
  __syncthreads();
  if (threadIdx.x < 32) {
    const int trt = threadIdx.x >> 4;   // row tile
    const int m   = threadIdx.x & 15;   // row within tile
    float tot = 0.f;
    #pragma unroll
    for (int p4 = 0; p4 < 4; ++p4)      // combine the 4 column phases
      tot += lds[(p4 * 2 + trt) * 16 + m];
    logz[blockIdx.x * 32 + threadIdx.x] = __logf(tot);
  }
}

// ---------------------------------------------------------------------------
// Kernel 3: loss = mean(logz - diag)
// ---------------------------------------------------------------------------
__global__ void simcse_loss(const float* __restrict__ logz,
                            const float* __restrict__ diag,
                            float* __restrict__ out) {
  float part = 0.f;
  for (int i = threadIdx.x; i < SIMN; i += 256)
    part += logz[i] - diag[i];
  #pragma unroll
  for (int off = 16; off > 0; off >>= 1) part += __shfl_down(part, off);
  __shared__ float s[8];
  const int wave = threadIdx.x >> 5, lane = threadIdx.x & 31;
  if (lane == 0) s[wave] = part;
  __syncthreads();
  if (threadIdx.x == 0) {
    float tot = 0.f;
    #pragma unroll
    for (int w = 0; w < 8; ++w) tot += s[w];
    out[0] = tot / (float)SIMN;
  }
}

// ---------------------------------------------------------------------------
extern "C" void kernel_launch(void* const* d_in, const int* in_sizes, int n_in,
                              void* d_out, int out_size, void* d_ws, size_t ws_size,
                              hipStream_t stream) {
  const float* q = (const float*)d_in[0];
  const float* p = (const float*)d_in[1];

  const size_t nd = (size_t)SIMN * SIMD;
  char* ws = (char*)d_ws;
  __bf16* qhi = (__bf16*)ws;                 // 4 MB each plane
  __bf16* qlo = qhi + nd;
  __bf16* phi = qlo + nd;
  __bf16* plo = phi + nd;
  float*  diag = (float*)(plo + nd);         // 32 KB
  float*  logz = diag + SIMN;                // 32 KB

  simcse_norm_split<<<SIMN, 256, 0, stream>>>(q, p, qhi, qlo, phi, plo, diag);
  simcse_gemm_logz<<<SIMN / 32, 256, 0, stream>>>(qhi, qlo, phi, plo, logz);
  simcse_loss<<<1, 256, 0, stream>>>(logz, diag, (float*)d_out);
}
